// JNetwork_6090263626486
// MI455X (gfx1250) — matrix operand order repaired and checked by reading, not compiled, not run on hardware
//
#include <hip/hip_runtime.h>

typedef __attribute__((ext_vector_type(8)))  float  v8f;
typedef __attribute__((ext_vector_type(16))) float  v16f;
typedef __attribute__((ext_vector_type(8)))  __bf16 v8bf;
typedef __attribute__((ext_vector_type(16))) __bf16 v16bf;

#define S_DIM 256
#define R_DIM 8192
#define RC    128       // r-chunk staged in LDS per iteration (4 K=32 steps)
#define BT    32        // b-rows per workgroup (two 16-row WMMA tiles)
#define AB_PITCH 257    // 256 abundances + ones column

// Split 16 consecutive f32 into packed bf16 hi + bf16 residual lo.
// hi = RNE(x); lo = RNE(x - hi)  (x - hi is exact in f32)
__device__ inline void split16(const float* __restrict__ p, v16bf& h, v16bf& l) {
    v16f x = *(const v16f*)p;
    #pragma unroll
    for (int i = 0; i < 16; ++i) {
        float xi  = x[i];
        __bf16 hi = (__bf16)xi;
        h[i] = hi;
        l[i] = (__bf16)(xi - (float)hi);
    }
}

__device__ inline v16bf joinA(const __bf16* __restrict__ p) {
    // lane's A-frag: 16 consecutive K at p (elements)
    v8bf a0 = *(const v8bf*)(p);
    v8bf a1 = *(const v8bf*)(p + 8);
    return __builtin_shufflevector(a0, a1, 0,1,2,3,4,5,6,7,8,9,10,11,12,13,14,15);
}

__global__ __launch_bounds__(256)
void jnet_flux_gemm(const float* __restrict__ ab,     // (B,S)
                    const float* __restrict__ temp,   // (B)
                    const float* __restrict__ crr,    // (B)
                    const float* __restrict__ fr,     // (B)
                    const float* __restrict__ inc,    // (S,R) row-major
                    const float* __restrict__ alpha,  // (R)
                    const float* __restrict__ beta,   // (R)
                    const float* __restrict__ gamma_, // (R)
                    const int2*  __restrict__ rm,     // (R,2)
                    const int*   __restrict__ rtype,  // (R)
                    float* __restrict__ out)          // (B,S)
{
    __shared__ float  ab_lds[BT * AB_PITCH];    // 32 abundance rows + ones col
    __shared__ float4 binv[BT];                 // (logT, 1/T, cr, fuv) per row
    __shared__ __bf16 fh_lds[2][BT * RC];       // flux hi, double-buffered
    __shared__ __bf16 fl_lds[2][BT * RC];       // flux lo (residual)

    const int t  = threadIdx.x;        // 0..255 (8 wave32s)
    const int b0 = blockIdx.x * BT;    // 32-row b-slab

    // ---- stage abundances rows of this b-slab into LDS (+ ones column) ----
    #pragma unroll
    for (int j = 0; j < BT; ++j) {
        int idx = j * 256 + t;             // 0..8191
        int bi  = idx >> 8;                // 0..31
        int s   = idx & 255;
        ab_lds[bi * AB_PITCH + s] = ab[(size_t)(b0 + bi) * S_DIM + s];
    }
    if (t < BT) {
        ab_lds[t * AB_PITCH + S_DIM] = 1.0f;
        float T = temp[b0 + t];
        binv[t] = make_float4(__logf(T * (1.0f / 300.0f)), 1.0f / T,
                              crr[b0 + t], fr[b0 + t]);
    }
    __syncthreads();

    const int lane = t & 31;
    const int ln16 = lane & 15;        // M (A/C row) or N (B col) index
    const int lh   = lane >> 4;        // lane half selects K sub-range
    const int wv   = t >> 5;           // wave id 0..7, owns s cols [wv*32, wv*32+32)

    // B-frag base: lane = col N = s-row of incidence; lane half picks K 0-15 / 16-31
    const float* incP0 = inc + (size_t)(wv * 32 + ln16) * R_DIM + lh * 16;
    const float* incP1 = incP0 + (size_t)16 * R_DIM;

    v8f acc00 = {}, acc01 = {};        // b-tile 0 x {s-tile 0, s-tile 1}
    v8f acc10 = {}, acc11 = {};        // b-tile 1 x {s-tile 0, s-tile 1}

    for (int c = 0; c < R_DIM / RC; ++c) {
        const int r0 = c * RC;
        __bf16* fhb = fh_lds[c & 1];
        __bf16* flb = fl_lds[c & 1];

        // ---- produce: flux[b0..b0+31][r0..r0+RC) as bf16 hi/lo in LDS ----
        #pragma unroll
        for (int j = 0; j < 16; ++j) {
            int idx = j * 256 + t;         // 0..4095
            int bi  = idx >> 7;            // 0..31 (wave-uniform)
            int rr  = idx & (RC - 1);
            int r   = r0 + rr;

            float al = alpha[r];
            float be = beta[r];
            float ga = gamma_[r];
            int   ty = rtype[r];
            int2  ii = rm[r];
            float4 bv = binv[bi];          // broadcast LDS read

            // branchless: k = alpha * mult * exp(arg)
            float arg = (ty == 0) ? fmaf(be, bv.x, -ga * bv.y)
                                  : ((ty == 2) ? -ga : 0.0f);
            float mul = (ty == 1) ? bv.z : ((ty == 2) ? bv.w : 1.0f);
            float k   = al * mul * __expf(arg);

            float a0 = ab_lds[bi * AB_PITCH + ii.x];
            float a1 = ab_lds[bi * AB_PITCH + ii.y];
            float fx = k * a0 * a1;

            __bf16 hv = (__bf16)fx;
            fhb[bi * RC + rr] = hv;
            flb[bi * RC + rr] = (__bf16)(fx - (float)hv);
        }
        __syncthreads();   // single barrier; double buffer makes this sufficient

        // ---- consume: compensated bf16x3 WMMAs over the chunk ----
        // A-frag (16x32 bf16): lane row M=ln16; lanes 0-15 hold K{0-15},
        // lanes 16-31 hold K{16-31} -> two ds_load_b128 per matrix.
        const __bf16* Ah0 = fhb + ln16 * RC + lh * 16;
        const __bf16* Al0 = flb + ln16 * RC + lh * 16;
        const __bf16* Ah1 = Ah0 + 16 * RC;      // b-tile 1 rows
        const __bf16* Al1 = Al0 + 16 * RC;
        const float*  i0p = incP0 + r0;
        const float*  i1p = incP1 + r0;

        if (c + 1 < R_DIM / RC) {          // warm next chunk of incidence
            __builtin_prefetch(i0p + RC, 0, 3);
            __builtin_prefetch(i1p + RC, 0, 3);
        }

        #pragma unroll
        for (int k32 = 0; k32 < RC / 32; ++k32) {
            v16bf ah0 = joinA(Ah0 + 32 * k32);
            v16bf al0 = joinA(Al0 + 32 * k32);
            v16bf ah1 = joinA(Ah1 + 32 * k32);
            v16bf al1 = joinA(Al1 + 32 * k32);

            v16bf bh, bl;
            split16(i0p + 32 * k32, bh, bl);             // s-tile 0
            acc00 = __builtin_amdgcn_wmma_f32_16x16x32_bf16(
                        false, ah0, false, bh, (short)0, acc00, false, false);
            acc10 = __builtin_amdgcn_wmma_f32_16x16x32_bf16(
                        false, ah1, false, bh, (short)0, acc10, false, false);
            acc00 = __builtin_amdgcn_wmma_f32_16x16x32_bf16(
                        false, ah0, false, bl, (short)0, acc00, false, false);
            acc10 = __builtin_amdgcn_wmma_f32_16x16x32_bf16(
                        false, ah1, false, bl, (short)0, acc10, false, false);
            acc00 = __builtin_amdgcn_wmma_f32_16x16x32_bf16(
                        false, al0, false, bh, (short)0, acc00, false, false);
            acc10 = __builtin_amdgcn_wmma_f32_16x16x32_bf16(
                        false, al1, false, bh, (short)0, acc10, false, false);

            split16(i1p + 32 * k32, bh, bl);             // s-tile 1
            acc01 = __builtin_amdgcn_wmma_f32_16x16x32_bf16(
                        false, ah0, false, bh, (short)0, acc01, false, false);
            acc11 = __builtin_amdgcn_wmma_f32_16x16x32_bf16(
                        false, ah1, false, bh, (short)0, acc11, false, false);
            acc01 = __builtin_amdgcn_wmma_f32_16x16x32_bf16(
                        false, ah0, false, bl, (short)0, acc01, false, false);
            acc11 = __builtin_amdgcn_wmma_f32_16x16x32_bf16(
                        false, ah1, false, bl, (short)0, acc11, false, false);
            acc01 = __builtin_amdgcn_wmma_f32_16x16x32_bf16(
                        false, al0, false, bh, (short)0, acc01, false, false);
            acc11 = __builtin_amdgcn_wmma_f32_16x16x32_bf16(
                        false, al1, false, bh, (short)0, acc11, false, false);
        }
    }

    // ---- writeback: C/D layout VGPR i, lane -> M = i + 8*lh, N = ln16 ----
    float* o = out + (size_t)(b0 + 8 * lh) * S_DIM + wv * 32 + ln16;
    #pragma unroll
    for (int i = 0; i < 8; ++i) {
        o[(size_t)i * S_DIM]             = acc00[i];
        o[(size_t)i * S_DIM + 16]        = acc01[i];
        o[(size_t)(i + 16) * S_DIM]      = acc10[i];
        o[(size_t)(i + 16) * S_DIM + 16] = acc11[i];
    }
}

extern "C" void kernel_launch(void* const* d_in, const int* in_sizes, int n_in,
                              void* d_out, int out_size, void* d_ws, size_t ws_size,
                              hipStream_t stream) {
    // setup_inputs order:
    // 0 time, 1 abundances, 2 temperature, 3 cr_rate, 4 fuv_rate,
    // 5 incidence, 6 alpha, 7 beta, 8 gamma, 9 reactant_multipliers, 10 rtype
    const float* ab   = (const float*)d_in[1];
    const float* temp = (const float*)d_in[2];
    const float* crr  = (const float*)d_in[3];
    const float* fr   = (const float*)d_in[4];
    const float* inc  = (const float*)d_in[5];
    const float* al   = (const float*)d_in[6];
    const float* be   = (const float*)d_in[7];
    const float* ga   = (const float*)d_in[8];
    const int2*  rmi  = (const int2*)d_in[9];
    const int*   rt   = (const int*)d_in[10];
    float* outp = (float*)d_out;

    const int B = in_sizes[2];            // temperature length = 8192
    dim3 grid(B / BT), block(256);
    jnet_flux_gemm<<<grid, block, 0, stream>>>(ab, temp, crr, fr, inc,
                                               al, be, ga, rmi, rt, outp);
}
